// _LSTMOneMany_3289944948986
// MI455X (gfx1250) — matrix operand order repaired and checked by reading, not compiled, not run on hardware
//
#include <hip/hip_runtime.h>
#include <hip/hip_bf16.h>
#include <math.h>

// ---------------------------------------------------------------------------
// GRU decoder (B=512, IN=1024, H=2048, L=2, O=512, T=32) for MI455X / gfx1250.
//
// All GEMMs:  A[M][K] @ W[N][K]^T + bias, dims multiples of the 128x128 block
// tile and 16x16x32 bf16 WMMA shape. Weights + activations are converted to
// bf16 ONCE per launch (~90MB ws, L2-resident across the 33 sequential
// steps), so the GEMM inner loop is a pure byte copy global->LDS -- done with
// GLOBAL_LOAD_ASYNC_TO_LDS_B128 (ASYNCcnt) double-buffered under the WMMAs.
// ---------------------------------------------------------------------------

#define BB  512
#define INF 1024
#define HH  2048
#define LL  2
#define OO  512
#define TT  32

typedef __attribute__((ext_vector_type(8)))  float  v8f;
typedef __attribute__((ext_vector_type(8)))  __bf16 v8bf;
typedef __attribute__((ext_vector_type(16))) __bf16 v16bf;
typedef __attribute__((ext_vector_type(4)))  int    i32x4;

typedef __attribute__((address_space(1))) i32x4* gas_b128;   // global src
typedef __attribute__((address_space(3))) i32x4* las_b128;   // LDS dst

#if defined(__has_builtin)
#if __has_builtin(__builtin_amdgcn_global_load_async_to_lds_b128) && \
    __has_builtin(__builtin_amdgcn_s_wait_asynccnt)
#define USE_ASYNC 1
#endif
#endif
#ifndef USE_ASYNC
#define USE_ASYNC 0
#endif

// fp32 -> bf16 bits, round-to-nearest-even.
__device__ __forceinline__ unsigned short f32_to_bf16_bits(float f) {
    unsigned u = __builtin_bit_cast(unsigned, f);
    u += 0x7FFFu + ((u >> 16) & 1u);
    return (unsigned short)(u >> 16);
}

// Epilogue modes
#define EPI_PLAIN 0   // out[row*ldout + col] = acc + bias[col]
#define EPI_EMB   1   // relu(acc+bias); row=b*L+l -> h f32 + bf16 mirrors
#define EPI_OUT   2   // d_out[row][t][col] f32 ; bf16 feedback xnext

// ---------------------------------------------------------------------------
// WMMA GEMM: 256 threads = 8 waves, block tile 128(M) x 128(N), waves 4x2,
// wave tile 32x64 = 2x4 WMMA tiles, K stepped by 32. LDS tiles double-
// buffered, rows padded to 40 bf16 (80B) to spread banks; b128 alignment
// holds (80*r + 16*q is 16B aligned).
// ---------------------------------------------------------------------------
template <int EPI>
__global__ __launch_bounds__(256) void gemm_wmma_bf16(
    const unsigned short* __restrict__ A,   // [M][K] bf16 bits
    const unsigned short* __restrict__ W,   // [N][K] bf16 bits (B^T)
    const float* __restrict__ bias,         // [N]
    int K, int ldout,
    float* __restrict__ out, unsigned short* __restrict__ out2, int t)
{
    constexpr int LDT = 40;
    __shared__ unsigned short As[2][128 * LDT];
    __shared__ unsigned short Bs[2][128 * LDT];

    const int tid  = threadIdx.x;
    const int lane = tid & 31;
    const int wave = tid >> 5;
    const int wm   = wave >> 1;          // 0..3 -> M offset wm*32
    const int wn   = wave & 1;           // 0..1 -> N offset wn*64
    const int half = lane >> 4;
    const int l16  = lane & 15;

    const int m0b = blockIdx.y * 128;
    const int n0b = blockIdx.x * 128;

    // Stage one 128x32 bf16 tile: 512 16B chunks, 2 per thread.
    auto stage = [&](unsigned short* dst, const unsigned short* src,
                     int row0, int k0) {
#pragma unroll
        for (int p = 0; p < 2; ++p) {
            const int c = tid + p * 256;          // chunk id 0..511
            const int r = c >> 2;                 // row 0..127
            const int q = (c & 3) * 8;            // k sub-offset (elements)
            const unsigned short* g = src + (size_t)(row0 + r) * K + k0 + q;
            unsigned short* l = dst + r * LDT + q;
#if USE_ASYNC
            __builtin_amdgcn_global_load_async_to_lds_b128(
                (gas_b128)(void*)g, (las_b128)(void*)l,
                /*offset=*/0, /*cpol=*/0);
#else
            *(uint4*)l = *(const uint4*)g;
#endif
        }
    };

    v8f acc[2][4];
#pragma unroll
    for (int i = 0; i < 2; ++i)
#pragma unroll
        for (int j = 0; j < 4; ++j) acc[i][j] = (v8f)0.0f;

    const int ksteps = K >> 5;            // K / 32
    stage(As[0], A, m0b, 0);
    stage(Bs[0], W, n0b, 0);

    for (int kt = 0; kt < ksteps; ++kt) {
        const int cur = kt & 1;
        if (kt + 1 < ksteps) {            // prefetch next tile into other buf
            stage(As[cur ^ 1], A, m0b, (kt + 1) << 5);
            stage(Bs[cur ^ 1], W, n0b, (kt + 1) << 5);
#if USE_ASYNC
            __builtin_amdgcn_s_wait_asynccnt(4);   // tile kt has landed
#endif
        } else {
#if USE_ASYNC
            __builtin_amdgcn_s_wait_asynccnt(0);
#endif
        }
        __syncthreads();

        // A 16x32 frag: lanes 0-15 row M=lane hold K 0..7 & 16..23,
        //               lanes16-31 row M=lane-16 hold K 8..15 & 24..31.
        v16bf af[2];
#pragma unroll
        for (int ti = 0; ti < 2; ++ti) {
            const int ar  = wm * 32 + ti * 16 + l16;
            const int akb = half * 8;
            v8bf lo = *(const v8bf*)&As[cur][ar * LDT + akb];
            v8bf hi = *(const v8bf*)&As[cur][ar * LDT + akb + 16];
            af[ti] = __builtin_shufflevector(lo, hi,
                     0,1,2,3,4,5,6,7,8,9,10,11,12,13,14,15);
        }
        // B 32x16 frag: lane holds column N=lane%16, K 8h..8h+7 & 16+8h..16+8h+7
        v16bf bf[4];
#pragma unroll
        for (int tj = 0; tj < 4; ++tj) {
            const int bc  = wn * 64 + tj * 16 + l16;
            const int bkb = half * 8;
            v8bf lo = *(const v8bf*)&Bs[cur][bc * LDT + bkb];
            v8bf hi = *(const v8bf*)&Bs[cur][bc * LDT + bkb + 16];
            bf[tj] = __builtin_shufflevector(lo, hi,
                     0,1,2,3,4,5,6,7,8,9,10,11,12,13,14,15);
        }

#pragma unroll
        for (int ti = 0; ti < 2; ++ti)
#pragma unroll
            for (int tj = 0; tj < 4; ++tj)
                acc[ti][tj] = __builtin_amdgcn_wmma_f32_16x16x32_bf16(
                    false, af[ti], false, bf[tj],
                    (short)0, acc[ti][tj], false, false);
        __syncthreads();
    }

    // Epilogue: C f32 16x16 -> element v of a lane is (M=v+8*half, N=lane%16)
#pragma unroll
    for (int ti = 0; ti < 2; ++ti) {
#pragma unroll
        for (int tj = 0; tj < 4; ++tj) {
            const int colg = n0b + wn * 64 + tj * 16 + l16;
            const float bv = bias[colg];
#pragma unroll
            for (int v = 0; v < 8; ++v) {
                const int rowg = m0b + wm * 32 + ti * 16 + half * 8 + v;
                float val = acc[ti][tj][v] + bv;
                if (EPI == EPI_PLAIN) {
                    out[(size_t)rowg * ldout + colg] = val;
                } else if (EPI == EPI_EMB) {
                    val = fmaxf(val, 0.0f);
                    const int b = rowg >> 1, l = rowg & 1;   // row = b*L + l
                    const size_t idx = (size_t)l * BB * HH + (size_t)b * HH + colg;
                    out[idx]  = val;
                    out2[idx] = f32_to_bf16_bits(val);
                } else {  // EPI_OUT
                    out[(size_t)rowg * (TT * OO) + (size_t)t * OO + colg] = val;
                    out2[(size_t)rowg * OO + colg] = f32_to_bf16_bits(val);
                }
            }
        }
    }
}

// ---------------------------------------------------------------------------
// GRU gate math (PyTorch order r,z,n); updates fp32 h and its bf16 mirror.
// ---------------------------------------------------------------------------
__global__ __launch_bounds__(256) void gru_gates(
    const float* __restrict__ gx, const float* __restrict__ gh,
    float* __restrict__ h, unsigned short* __restrict__ hbf)
{
    const int i = blockIdx.x * blockDim.x + threadIdx.x;   // over B*H
    if (i >= BB * HH) return;
    const int b = i / HH, j = i % HH;
    const size_t o = (size_t)b * (3 * HH) + j;
    const float r = 1.0f / (1.0f + __expf(-(gx[o] + gh[o])));
    const float z = 1.0f / (1.0f + __expf(-(gx[o + HH] + gh[o + HH])));
    const float n = tanhf(gx[o + 2 * HH] + r * gh[o + 2 * HH]);
    const float hn = (1.0f - z) * n + z * h[i];
    h[i]   = hn;
    hbf[i] = f32_to_bf16_bits(hn);
}

__global__ __launch_bounds__(256) void cvt_f32_bf16(
    const float* __restrict__ src, unsigned short* __restrict__ dst, int n)
{
    const int i = blockIdx.x * blockDim.x + threadIdx.x;
    if (i < n) dst[i] = f32_to_bf16_bits(src[i]);
}

__global__ __launch_bounds__(256) void zero_u16(unsigned short* __restrict__ p, int n) {
    const int i = blockIdx.x * blockDim.x + threadIdx.x;
    if (i < n) p[i] = 0;   // bf16 bits 0 == 0.0f
}

// ---------------------------------------------------------------------------
// Host-side sequencing (graph-capture safe: only stream launches).
// ---------------------------------------------------------------------------
extern "C" void kernel_launch(void* const* d_in, const int* in_sizes, int n_in,
                              void* d_out, int out_size, void* d_ws, size_t ws_size,
                              hipStream_t stream) {
    (void)in_sizes; (void)n_in; (void)out_size; (void)ws_size;
    const float* x     = (const float*)d_in[0];   // [B, L, IN]
    const float* emb_W = (const float*)d_in[1];   // [H, IN]
    const float* emb_b = (const float*)d_in[2];
    const float* Wih0  = (const float*)d_in[3];   // [3H, O]
    const float* Whh0  = (const float*)d_in[4];   // [3H, H]
    const float* bih0  = (const float*)d_in[5];
    const float* bhh0  = (const float*)d_in[6];
    const float* Wih1  = (const float*)d_in[7];   // [3H, H]
    const float* Whh1  = (const float*)d_in[8];   // [3H, H]
    const float* bih1  = (const float*)d_in[9];
    const float* bhh1  = (const float*)d_in[10];
    const float* out_W = (const float*)d_in[11];  // [O, H]
    const float* out_b = (const float*)d_in[12];
    float* out = (float*)d_out;                   // [B, T, O]

    // ---- workspace carve-up (~124 MB: 33MB f32 state + 91MB bf16) ----
    char* p = (char*)d_ws;
    auto carve = [&](size_t bytes) {
        char* r = p; p += (bytes + 255) & ~(size_t)255; return r;
    };
    float* hbuf = (float*)carve(sizeof(float) * 2 * BB * HH);   // [2][B][H]
    float* gx   = (float*)carve(sizeof(float) * BB * 3 * HH);
    float* gh   = (float*)carve(sizeof(float) * BB * 3 * HH);
    unsigned short* hbf    = (unsigned short*)carve(2ull * 2 * BB * HH);
    unsigned short* xbf    = (unsigned short*)carve(2ull * BB * LL * INF);
    unsigned short* xcurbf = (unsigned short*)carve(2ull * BB * OO);
    unsigned short* embWbf = (unsigned short*)carve(2ull * HH * INF);
    unsigned short* Wih0bf = (unsigned short*)carve(2ull * 3 * HH * OO);
    unsigned short* Whh0bf = (unsigned short*)carve(2ull * 3 * HH * HH);
    unsigned short* Wih1bf = (unsigned short*)carve(2ull * 3 * HH * HH);
    unsigned short* Whh1bf = (unsigned short*)carve(2ull * 3 * HH * HH);
    unsigned short* outWbf = (unsigned short*)carve(2ull * OO * HH);
    float* h0 = hbuf;
    float* h1 = hbuf + (size_t)BB * HH;
    unsigned short* h0bf = hbf;
    unsigned short* h1bf = hbf + (size_t)BB * HH;

    const dim3 blk(256);
    const dim3 gGate(3 * HH / 128, BB / 128);     // 48 x 4
    const dim3 gEmb(HH / 128, (BB * LL) / 128);   // 16 x 8
    const dim3 gOut(OO / 128, BB / 128);          //  4 x 4
    const int  nGates = (BB * HH + 255) / 256;

    // ---- one-time bf16 conversion of weights + input (L2-resident after) --
    auto cvt = [&](const float* s, unsigned short* d, int n) {
        cvt_f32_bf16<<<(n + 255) / 256, blk, 0, stream>>>(s, d, n);
    };
    cvt(x,     xbf,    BB * LL * INF);
    cvt(emb_W, embWbf, HH * INF);
    cvt(Wih0,  Wih0bf, 3 * HH * OO);
    cvt(Whh0,  Whh0bf, 3 * HH * HH);
    cvt(Wih1,  Wih1bf, 3 * HH * HH);
    cvt(Whh1,  Whh1bf, 3 * HH * HH);
    cvt(out_W, outWbf, OO * HH);
    zero_u16<<<(BB * OO + 255) / 256, blk, 0, stream>>>(xcurbf, BB * OO);

    // ---- embedding: h = relu(x @ emb_W^T + b), (B,L)->(L,B) row remap ----
    gemm_wmma_bf16<EPI_EMB><<<gEmb, blk, 0, stream>>>(
        xbf, embWbf, emb_b, INF, 0, hbuf, hbf, 0);

    // ---- 33 GRU steps: s=0,1 warm-up (zero input); out-proj t=s-1 ----
    for (int s = 0; s < TT + 1; ++s) {
        gemm_wmma_bf16<EPI_PLAIN><<<gGate, blk, 0, stream>>>(
            xcurbf, Wih0bf, bih0, OO, 3 * HH, gx, nullptr, 0);
        gemm_wmma_bf16<EPI_PLAIN><<<gGate, blk, 0, stream>>>(
            h0bf, Whh0bf, bhh0, HH, 3 * HH, gh, nullptr, 0);
        gru_gates<<<nGates, blk, 0, stream>>>(gx, gh, h0, h0bf);
        gemm_wmma_bf16<EPI_PLAIN><<<gGate, blk, 0, stream>>>(
            h0bf, Wih1bf, bih1, HH, 3 * HH, gx, nullptr, 0);
        gemm_wmma_bf16<EPI_PLAIN><<<gGate, blk, 0, stream>>>(
            h1bf, Whh1bf, bhh1, HH, 3 * HH, gh, nullptr, 0);
        gru_gates<<<nGates, blk, 0, stream>>>(gx, gh, h1, h1bf);
        if (s >= 1) {
            gemm_wmma_bf16<EPI_OUT><<<gOut, blk, 0, stream>>>(
                h1bf, outWbf, out_b, HH, 0, out, xcurbf, s - 1);
        }
    }
}
